// LePELinearAttention_62302795595837
// MI455X (gfx1250) — matrix operand order repaired and checked by reading, not compiled, not run on hardware
//
#include <hip/hip_runtime.h>
#include <hip/hip_bf16.h>
#include <math.h>

// ---------------------------------------------------------------------------
// LePE linear attention for MI455X (gfx1250, wave32, WMMA).
// Shapes: B=4, L=4096, C=1024, H=16, D=64, BH=64, M=B*L=16384.
// ---------------------------------------------------------------------------

typedef __attribute__((ext_vector_type(16))) _Float16 v16h;
typedef __attribute__((ext_vector_type(8)))  _Float16 v8h;
typedef __attribute__((ext_vector_type(4)))  _Float16 v4h;
typedef __attribute__((ext_vector_type(8)))  float    v8f;

#define LSEQ   4096
#define DHEAD  64
#define NHEAD  16
#define BATCH  4
#define BH     (BATCH * NHEAD)           // 64
#define MROWS  (BATCH * LSEQ)            // 16384
#define KDIM   1024
#define NQKV   3072                      // 2048 qk + 1024 v
#define PLANE  (LSEQ * DHEAD)            // 262144 elements per head plane

// exp(x) via single hardware v_exp_f32 (no libcall -> no VGPR spill at calls)
__device__ __forceinline__ float fast_exp(float x) {
  return __builtin_amdgcn_exp2f(x * 1.44269504088896340736f);
}

// ---------------- WMMA fragment loaders (f16, 16x16x32) --------------------
// A (16xK, row-major, K contiguous): lane m = lane&15, half picks +16 K
// offset -> two contiguous 16B chunks at k and k+8.
__device__ __forceinline__ v16h load_a_frag(const _Float16* __restrict__ base,
                                            int lda, int m0, int k0) {
  int lane = threadIdx.x & 31;
  int m    = m0 + (lane & 15);
  int koff = k0 + ((lane >> 4) << 4);
  const _Float16* p = base + m * lda + koff;
  union { v16h v; v8h h[2]; } u;
  u.h[0] = *(const v8h*)(p);
  u.h[1] = *(const v8h*)(p + 8);
  return u.v;
}
// B stored transposed (N rows, K contiguous): lane n = lane&15, half picks
// +16 K offset -> one contiguous 32B chunk.
__device__ __forceinline__ v16h load_b_frag(const _Float16* __restrict__ baseT,
                                            int ldb, int n0, int k0) {
  int lane = threadIdx.x & 31;
  int n    = n0 + (lane & 15);
  int koff = k0 + ((lane >> 4) << 4);
  return *(const v16h*)(baseT + n * ldb + koff);
}

// ---------------- small helper kernels -------------------------------------
// n must be a multiple of 4 (true for all uses here).
__global__ void pack_h_kernel(const float* __restrict__ src,
                              _Float16* __restrict__ dst, int n4) {
  int i = blockIdx.x * blockDim.x + threadIdx.x;
  if (i < n4) {
    float4 f = ((const float4*)src)[i];
    v4h h = { (_Float16)f.x, (_Float16)f.y, (_Float16)f.z, (_Float16)f.w };
    ((v4h*)dst)[i] = h;
  }
}

// (BH, L, D) f32  ->  (BH, D, L) f16
__global__ void packT_h_kernel(const float* __restrict__ src,
                               _Float16* __restrict__ dst) {
  int idx = blockIdx.x * 256 + threadIdx.x;     // 64*262144 total
  int bh  = idx >> 18;
  int rem = idx & (PLANE - 1);
  int d   = rem >> 12;
  int l   = rem & (LSEQ - 1);
  dst[bh * PLANE + d * LSEQ + l] = (_Float16)src[bh * PLANE + l * DHEAD + d];
}

// ---------------- fused QKV projection GEMM --------------------------------
// y[m,n] = sum_k xh[m,k] * wh[n,k]  (+bias, +elu+1 for q/k)
// Each wave computes a 64x64 C tile; 8 waves/block cover 512 rows.
__global__ __launch_bounds__(256) void gemm_qkv_kernel(
    const _Float16* __restrict__ xh, const _Float16* __restrict__ wh,
    const float* __restrict__ b_qk, const float* __restrict__ b_v,
    float* __restrict__ qraw, float* __restrict__ kraw,
    _Float16* __restrict__ vt) {
  int wave = threadIdx.x >> 5;
  int lane = threadIdx.x & 31;
  int n0 = blockIdx.x * 64;
  int m0 = (blockIdx.y * 8 + wave) * 64;

  v8f acc[4][4];
  for (int i = 0; i < 4; ++i)
    for (int j = 0; j < 4; ++j)
      acc[i][j] = (v8f){0.f,0.f,0.f,0.f,0.f,0.f,0.f,0.f};

  for (int k0 = 0; k0 < KDIM; k0 += 32) {
    v16h a[4], b[4];
    for (int i = 0; i < 4; ++i) a[i] = load_a_frag(xh, KDIM, m0 + 16*i, k0);
    for (int j = 0; j < 4; ++j) b[j] = load_b_frag(wh, KDIM, n0 + 16*j, k0);
    for (int i = 0; i < 4; ++i)
      for (int j = 0; j < 4; ++j)
        acc[i][j] = __builtin_amdgcn_wmma_f32_16x16x32_f16(
            false, a[i], false, b[j], (short)0, acc[i][j], false, false);
  }

  // Epilogue. Region (q/k/v) is uniform for the whole block: n0 % 64 == 0 and
  // region boundaries (1024, 2048) are multiples of 64. A 64-row tile never
  // crosses the L=4096 batch boundary, so bb/l0 are wave-uniform too.
  int lh = lane >> 4, ll = lane & 15;
  int bb = m0 >> 12;
  int l0 = m0 & (LSEQ - 1);
  int region = n0 >> 10;                       // 0=q, 1=k, 2=v

  if (region < 2) {
    float* __restrict__ dst = (region == 0) ? qraw : kraw;
    for (int j = 0; j < 4; ++j) {
      int n = n0 + 16*j + ll;
      float bias = b_qk[n];
      int h = (n >> 6) & 15;
      int d = n & 63;
      float* colp = dst + (size_t)(bb * NHEAD + h) * PLANE + d;
      for (int i = 0; i < 4; ++i) {
        float* p = colp + (size_t)(l0 + 16*i + 8*lh) * DHEAD;
        for (int r = 0; r < 8; ++r) {
          float vq = acc[i][j][r] + bias;
          vq = (vq > 0.f) ? (vq + 1.f) : fast_exp(vq);   // elu(x)+1
          p[r * DHEAD] = vq;                             // imm offset r*256B
        }
      }
    }
  } else {
    for (int j = 0; j < 4; ++j) {
      int n2 = (n0 - 2048) + 16*j + ll;
      float bias = b_v[n2];
      int h = n2 >> 6;
      int d = n2 & 63;
      // V written directly transposed f16: (BH, D, L)
      _Float16* colp = vt + (size_t)(bb * NHEAD + h) * PLANE + (size_t)d * LSEQ;
      for (int i = 0; i < 4; ++i) {
        _Float16* p = colp + (l0 + 16*i + 8*lh);
        for (int r = 0; r < 8; ++r)
          p[r] = (_Float16)(acc[i][j][r] + bias);        // imm offset r*2B
      }
    }
  }
}

// ---------------- LePE: a += depthwise-conv(flat-reinterp(a)) --------------
// For output (l,d): m = d*L + l over the flat head plane f;
// add w[d,0]*f(m-1) + w[d,1]*f(m) + w[d,2]*f(m+1) + bias[d]
// (zero at l==0 / l==L-1 row edges of the reinterpreted array).
__global__ __launch_bounds__(256) void lepe_kernel(
    const float* __restrict__ raw, float* __restrict__ outp,
    const float* __restrict__ w, const float* __restrict__ bias) {
  int bh = blockIdx.x;                    // 64
  int l0 = blockIdx.y * 64;               // 64 tiles over L
  const float* plane = raw  + (size_t)bh * PLANE;
  float*       op    = outp + (size_t)bh * PLANE;
  int l     = l0 + (threadIdx.x & 63);
  int dbase = threadIdx.x >> 6;           // 0..3
  for (int it = 0; it < 16; ++it) {
    int d = dbase * 16 + it;
    int m = d * LSEQ + l;
    float c  = plane[m];
    float lf = (l == 0)        ? 0.f : plane[m - 1];
    float rf = (l == LSEQ - 1) ? 0.f : plane[m + 1];
    float conv = w[d*3+0]*lf + w[d*3+1]*c + w[d*3+2]*rf + bias[d];
    op[l * DHEAD + d] = plane[l * DHEAD + d] + conv;
  }
}

// ---------------- k_mean over L --------------------------------------------
__global__ __launch_bounds__(64) void kmean_kernel(
    const float* __restrict__ kp, float* __restrict__ km) {
  int bh = blockIdx.x;
  int d  = threadIdx.x;
  const float* p = kp + (size_t)bh * PLANE + d;
  float s = 0.f;
  for (int l = 0; l < LSEQ; ++l) s += p[l * DHEAD];
  km[bh * DHEAD + d] = s * (1.0f / (float)LSEQ);
}

// ---------------- z = 1/(q . k_mean + 1e-6) --------------------------------
__global__ __launch_bounds__(256) void z_kernel(
    const float* __restrict__ qp, const float* __restrict__ km,
    float* __restrict__ z) {
  int idx = blockIdx.x * 256 + threadIdx.x;       // BH*L = 262144
  int bh = idx >> 12;
  int l  = idx & (LSEQ - 1);
  const float* q = qp + (size_t)bh * PLANE + l * DHEAD;
  const float* m = km + bh * DHEAD;
  float s = 0.f;
  for (int d = 0; d < DHEAD; ++d) s += q[d] * m[d];
  z[idx] = 1.0f / (s + 1e-6f);
}

// ---------------- kv = K^T V (K-sliced partials, WMMA) ---------------------
__global__ __launch_bounds__(32) void kv_gemm_kernel(
    const _Float16* __restrict__ kpt, const _Float16* __restrict__ vt,
    float* __restrict__ part) {
  int head  = blockIdx.x;                 // 64
  int slice = blockIdx.y;                 // 8
  const _Float16* A = kpt + (size_t)head * PLANE;   // (D=64 x L), K contiguous
  const _Float16* B = vt  + (size_t)head * PLANE;   // (E=64 x L), K contiguous
  int kstart = slice * 512;

  v8f acc[4][4];
  for (int i = 0; i < 4; ++i)
    for (int j = 0; j < 4; ++j)
      acc[i][j] = (v8f){0.f,0.f,0.f,0.f,0.f,0.f,0.f,0.f};

  for (int k0 = kstart; k0 < kstart + 512; k0 += 32) {
    v16h a[4], b[4];
    for (int i = 0; i < 4; ++i) a[i] = load_a_frag(A, LSEQ, 16*i, k0);
    for (int j = 0; j < 4; ++j) b[j] = load_b_frag(B, LSEQ, 16*j, k0);
    for (int i = 0; i < 4; ++i)
      for (int j = 0; j < 4; ++j)
        acc[i][j] = __builtin_amdgcn_wmma_f32_16x16x32_f16(
            false, a[i], false, b[j], (short)0, acc[i][j], false, false);
  }

  int lane = threadIdx.x & 31;
  int lh = lane >> 4, ll = lane & 15;
  float* dst = part + ((size_t)slice * BH + head) * (DHEAD * DHEAD);
  for (int i = 0; i < 4; ++i)
    for (int r = 0; r < 8; ++r) {
      float* rowp = dst + (16*i + r + 8*lh) * DHEAD + ll;
      for (int j = 0; j < 4; ++j)
        rowp[16*j] = acc[i][j][r];                  // imm offset j*64B
    }
}

// Deterministic slice reduction; writes kv transposed (E rows, D contiguous)
// as f16 for the final GEMM's B operand; applies /L.
__global__ __launch_bounds__(256) void kv_reduce_kernel(
    const float* __restrict__ part, _Float16* __restrict__ kvt) {
  int idx  = blockIdx.x * 256 + threadIdx.x;     // BH*64*64 = 262144
  int head = idx >> 12;
  int d    = (idx >> 6) & 63;
  int e    = idx & 63;
  float s = 0.f;
  for (int sl = 0; sl < 8; ++sl)
    s += part[((size_t)sl * BH + head) * (DHEAD * DHEAD) + d * DHEAD + e];
  kvt[(size_t)head * (DHEAD * DHEAD) + e * DHEAD + d] =
      (_Float16)(s * (1.0f / (float)LSEQ));
}

// ---------------- out = (Q @ kv) * z  -> (B, L, H, D) ----------------------
__global__ __launch_bounds__(256) void out_gemm_kernel(
    const _Float16* __restrict__ qph, const _Float16* __restrict__ kvt,
    const float* __restrict__ z, float* __restrict__ out) {
  int wave = threadIdx.x >> 5;
  int lane = threadIdx.x & 31;
  int wg   = blockIdx.x * 8 + wave;       // 0..4095
  int head = wg >> 6;
  int m0   = (wg & 63) * 64;
  const _Float16* A = qph + (size_t)head * PLANE;          // (L x 64)
  const _Float16* B = kvt + (size_t)head * (DHEAD * DHEAD); // (64 x 64) e-major

  v8f acc[4][4];
  for (int i = 0; i < 4; ++i)
    for (int j = 0; j < 4; ++j)
      acc[i][j] = (v8f){0.f,0.f,0.f,0.f,0.f,0.f,0.f,0.f};

  for (int k0 = 0; k0 < DHEAD; k0 += 32) {
    v16h a[4], b[4];
    for (int i = 0; i < 4; ++i) a[i] = load_a_frag(A, DHEAD, m0 + 16*i, k0);
    for (int j = 0; j < 4; ++j) b[j] = load_b_frag(B, DHEAD, 16*j, k0);
    for (int i = 0; i < 4; ++i)
      for (int j = 0; j < 4; ++j)
        acc[i][j] = __builtin_amdgcn_wmma_f32_16x16x32_f16(
            false, a[i], false, b[j], (short)0, acc[i][j], false, false);
  }

  int lh = lane >> 4, ll = lane & 15;
  int bb = head >> 4;
  int h  = head & 15;
  const float* zp = z + head * LSEQ;
  for (int i = 0; i < 4; ++i) {
    int lrow = m0 + 16*i + 8*lh;
    for (int r = 0; r < 8; ++r) {
      int l = lrow + r;
      float zz = zp[l];
      float* rowp = out + (((size_t)bb * LSEQ + l) * NHEAD + h) * DHEAD + ll;
      for (int j = 0; j < 4; ++j)
        rowp[16*j] = acc[i][j][r] * zz;             // imm offset j*64B
    }
  }
}

// ---------------------------------------------------------------------------
extern "C" void kernel_launch(void* const* d_in, const int* in_sizes, int n_in,
                              void* d_out, int out_size, void* d_ws, size_t ws_size,
                              hipStream_t stream) {
  const float* x      = (const float*)d_in[0];   // (B,L,C)
  const float* w_qk   = (const float*)d_in[1];   // (2C,C)
  const float* b_qk   = (const float*)d_in[2];   // (2C)
  const float* w_v    = (const float*)d_in[3];   // (C,C)
  const float* b_v    = (const float*)d_in[4];   // (C)
  const float* w_lepe = (const float*)d_in[5];   // (D,1,3)
  const float* b_lepe = (const float*)d_in[6];   // (D)
  float* out = (float*)d_out;

  char* ws = (char*)d_ws;
  const size_t MB = 1ull << 20;
  // live ranges chosen so aliased regions never overlap in time
  _Float16* xh   = (_Float16*)(ws + 0);          // 32 MB   x as f16
  _Float16* wh   = (_Float16*)(ws + 32 * MB);    //  6 MB   [w_qk; w_v] f16
  float*    qraw = (float*)   (ws + 40 * MB);    // 64 MB   q pre-LePE
  float*    kraw = (float*)   (ws + 104 * MB);   // 64 MB   k pre-LePE
  _Float16* vt   = (_Float16*)(ws + 168 * MB);   // 32 MB   V^T f16 (BH,D,L)
  float*    qp   = (float*)   (ws + 200 * MB);   // 64 MB   q post-LePE
  float*    kp   = (float*)   (ws + 264 * MB);   // 64 MB   k post-LePE
  // aliases into dead regions:
  _Float16* qph  = (_Float16*)(ws + 40 * MB);        // 32 MB (over qraw)
  _Float16* kpt  = (_Float16*)(ws + 72 * MB);        // 32 MB (over qraw)
  float*    part = (float*)   (ws + 104 * MB);       //  8 MB (over kraw)
  _Float16* kvt  = (_Float16*)(ws + 112 * MB);       // .5 MB (over kraw)
  float*    km   = (float*)   (ws + 113 * MB);       // 16 KB (over kraw)
  float*    zbuf = (float*)   (ws + 114 * MB);       //  1 MB (over kraw)

  const int NX   = MROWS * KDIM;        // 16,777,216
  const int NWQK = 2048 * KDIM;         //  2,097,152
  const int NWV  = 1024 * KDIM;         //  1,048,576

  // 1) pack inputs to f16 (vectorized x4)
  pack_h_kernel<<<(NX/4 + 255) / 256, 256, 0, stream>>>(x, xh, NX/4);
  pack_h_kernel<<<(NWQK/4 + 255) / 256, 256, 0, stream>>>(w_qk, wh, NWQK/4);
  pack_h_kernel<<<(NWV/4 + 255) / 256, 256, 0, stream>>>(w_v, wh + NWQK, NWV/4);

  // 2) fused QKV projection (WMMA), elu+1 for q/k, V packed transposed
  gemm_qkv_kernel<<<dim3(NQKV / 64, MROWS / 512), 256, 0, stream>>>(
      xh, wh, b_qk, b_v, qraw, kraw, vt);

  // 3) LePE on q and k
  lepe_kernel<<<dim3(BH, LSEQ / 64), 256, 0, stream>>>(qraw, qp, w_lepe, b_lepe);
  lepe_kernel<<<dim3(BH, LSEQ / 64), 256, 0, stream>>>(kraw, kp, w_lepe, b_lepe);

  // 4) k_mean, f16 packs, z
  kmean_kernel<<<BH, 64, 0, stream>>>(kp, km);
  pack_h_kernel<<<(BH * PLANE / 4 + 255) / 256, 256, 0, stream>>>(qp, qph, BH * PLANE / 4);
  packT_h_kernel<<<(BH * PLANE) / 256, 256, 0, stream>>>(kp, kpt);
  z_kernel<<<(BH * LSEQ) / 256, 256, 0, stream>>>(qp, km, zbuf);

  // 5) kv = K^T V / L (WMMA, K-sliced partials + deterministic reduce)
  kv_gemm_kernel<<<dim3(BH, 8), 32, 0, stream>>>(kpt, vt, part);
  kv_reduce_kernel<<<(BH * DHEAD * DHEAD) / 256, 256, 0, stream>>>(part, kvt);

  // 6) out = (Q @ kv) * z   (WMMA)
  out_gemm_kernel<<<(BH * (LSEQ / 64)) / 8, 256, 0, stream>>>(qph, kvt, zbuf, out);
}
